// NeuroPhyloLSTM_52072183497130
// MI455X (gfx1250) — matrix operand order repaired and checked by reading, not compiled, not run on hardware
//
#include <hip/hip_runtime.h>

// ---------------------------------------------------------------------------
// Types / helpers
// ---------------------------------------------------------------------------
typedef _Float16 half8 __attribute__((ext_vector_type(8)));
typedef _Float16 v16h  __attribute__((ext_vector_type(16)));
typedef float    v8f   __attribute__((ext_vector_type(8)));

__device__ __forceinline__ v8f vzero8() {
  v8f z = {0.f, 0.f, 0.f, 0.f, 0.f, 0.f, 0.f, 0.f};
  return z;
}

__device__ __forceinline__ float sigm(float x) {
  return 1.0f / (1.0f + __expf(-x));
}

__device__ __forceinline__ v8f wmma_f16(v16h a, v16h b, v8f c) {
  // (neg_a, A, neg_b, B, c_mod, C, reuse_a, reuse_b)
  return __builtin_amdgcn_wmma_f32_16x16x32_f16(false, a, false, b, (short)0, c,
                                                false, false);
}

// A-fragment (16x32 f16, M x K) from row-major LDS buffer [32][ld] (halves).
// Per ISA: lane l holds row M=l&15; elements 0..7 = K kb..kb+7,
// elements 8..15 = K kb+16..kb+23, kb = (l>>4)*8 within the 32-wide k-tile.
__device__ __forceinline__ v16h afrag_lds(const _Float16* base, int ld,
                                          int row0, int col0, int kt) {
  const int lane = threadIdx.x & 31;
  const int r = row0 + (lane & 15);
  const int kb = col0 + kt * 32 + ((lane >> 4) << 3);
  const _Float16* p = base + r * ld + kb;
  half8 lo = *(const half8*)(p);
  half8 hi = *(const half8*)(p + 16);
  return __builtin_shufflevector(lo, hi, 0, 1, 2, 3, 4, 5, 6, 7, 8, 9, 10, 11,
                                 12, 13, 14, 15);
}

// B-fragment (32x16 f16, K x N): weights pre-packed tile-major, 512 halves per
// tile, each lane's 16 halves contiguous (32B) -> one coalesced load.
// Per ISA: lane l holds col N=l&15; elements e -> K = (l>>4)*8 + (e&7) + (e>>3)*16.
__device__ __forceinline__ v16h bfrag(const _Float16* Wt, int tile) {
  return ((const v16h*)Wt)[tile * 32 + (threadIdx.x & 31)];
}

// ---------------------------------------------------------------------------
// One LSTM cell step for the whole workgroup (16 waves, B=32, H=256).
// A = [x | h_prev] row-major in LDS ([32][LDA] halves, K = KT*32 valid cols).
// Wave w computes gate columns j in [16w, 16w+16) for all four gates.
// cfr (cell state fragments) live in VGPRs across the whole sequence.
// ---------------------------------------------------------------------------
template <int KT, int LDA>
__device__ __forceinline__ void lstm_cell_step(const _Float16* Abuf,
                                               const _Float16* Wt, float bi,
                                               float bf, float bg, float bo,
                                               v8f cfr[2], float hout[2][8]) {
  const int w = threadIdx.x >> 5;
  v8f acc[4][2];
#pragma unroll
  for (int g = 0; g < 4; ++g) {
    acc[g][0] = vzero8();
    acc[g][1] = vzero8();
  }
  for (int kt = 0; kt < KT; ++kt) {
    v16h a0 = afrag_lds(Abuf, LDA, 0, 0, kt);
    v16h a1 = afrag_lds(Abuf, LDA, 16, 0, kt);
#pragma unroll
    for (int g = 0; g < 4; ++g) {
      v16h b = bfrag(Wt, (g * 16 + w) * KT + kt);
      acc[g][0] = wmma_f16(a0, b, acc[g][0]);
      acc[g][1] = wmma_f16(a1, b, acc[g][1]);
    }
  }
#pragma unroll
  for (int mt = 0; mt < 2; ++mt) {
#pragma unroll
    for (int v = 0; v < 8; ++v) {
      float iv = sigm(acc[0][mt][v] + bi);
      float fv = sigm(acc[1][mt][v] + bf);
      float gv = tanhf(acc[2][mt][v] + bg);
      float ov = sigm(acc[3][mt][v] + bo);
      float cn = fv * cfr[mt][v] + iv * gv;
      cfr[mt][v] = cn;
      hout[mt][v] = ov * tanhf(cn);
    }
  }
}

// ---------------------------------------------------------------------------
// Persistent bidirectional-encoder layer kernel. blockIdx.x = direction.
// xsrc: [B*S][XC] f16 rows (time-major within batch row index b*256+t).
// ---------------------------------------------------------------------------
template <int KT, int LDA, int XC, bool SEQ_OUT, bool FIN_OUT>
__global__ void __launch_bounds__(512) lstm_enc_kernel(
    const _Float16* __restrict__ xsrc, const _Float16* __restrict__ WtF,
    const _Float16* __restrict__ WtB, const float* __restrict__ biasF,
    const float* __restrict__ biasB, _Float16* __restrict__ oseq,
    float* __restrict__ hfin, float* __restrict__ cfin) {
  constexpr int HOFF = XC;
  __shared__ _Float16 Abuf[32 * LDA];
  const int dir = blockIdx.x;
  const bool rev = (dir != 0);
  const _Float16* Wt = rev ? WtB : WtF;
  const float* bias = rev ? biasB : biasF;
  const int tid = threadIdx.x;
  const int lane = tid & 31;
  const int w = tid >> 5;
  const int lh = lane >> 4, jn = lane & 15;
  const int j = w * 16 + jn;
  const float bi = bias[j], bf = bias[256 + j];
  const float bg = bias[512 + j], bo = bias[768 + j];

  // h(-1) = 0
  for (int idx = tid; idx < 32 * 256; idx += 512)
    Abuf[(idx >> 8) * LDA + HOFF + (idx & 255)] = (_Float16)0.f;
  // x for first step
  {
    const int t0 = rev ? 255 : 0;
    const int r = tid >> 4;
    const int cb = (tid & 15) * (XC / 16);
#pragma unroll
    for (int c = 0; c < XC / 16; c += 8)
      *(half8*)&Abuf[r * LDA + cb + c] =
          *(const half8*)&xsrc[(r * 256 + t0) * XC + cb + c];
  }
  v8f cfr[2] = {vzero8(), vzero8()};
  __syncthreads();

  for (int t = 0; t < 256; ++t) {
    float hout[2][8];
    lstm_cell_step<KT, LDA>(Abuf, Wt, bi, bf, bg, bo, cfr, hout);
    __syncthreads();  // everyone done reading A
    const int tphys = rev ? (255 - t) : t;
#pragma unroll
    for (int mt = 0; mt < 2; ++mt)
#pragma unroll
      for (int v = 0; v < 8; ++v) {
        const int m = mt * 16 + lh * 8 + v;
        const _Float16 hh = (_Float16)hout[mt][v];
        Abuf[m * LDA + HOFF + j] = hh;
        if (SEQ_OUT) oseq[(m * 256 + tphys) * 512 + dir * 256 + j] = hh;
      }
    if (FIN_OUT) {
      if (t == 255) {
#pragma unroll
        for (int mt = 0; mt < 2; ++mt)
#pragma unroll
          for (int v = 0; v < 8; ++v) {
            const int m = mt * 16 + lh * 8 + v;
            hfin[(dir * 32 + m) * 256 + j] = hout[mt][v];
            cfin[(dir * 32 + m) * 256 + j] = cfr[mt][v];
          }
      }
    }
    if (t < 255) {  // stage x for next step
      const int tn = rev ? (254 - t) : (t + 1);
      const int r = tid >> 4;
      const int cb = (tid & 15) * (XC / 16);
#pragma unroll
      for (int c = 0; c < XC / 16; c += 8)
        *(half8*)&Abuf[r * LDA + cb + c] =
            *(const half8*)&xsrc[(r * 256 + tn) * XC + cb + c];
    }
    __syncthreads();
  }
}

// ---------------------------------------------------------------------------
// Bridge: dh = [h1f|h1b] @ hbW^T + hb_b ; dc likewise with cbW.
// ---------------------------------------------------------------------------
__global__ void __launch_bounds__(512) bridge_kernel(
    const float* __restrict__ hfin, const float* __restrict__ cfin,
    const _Float16* __restrict__ Whb, const _Float16* __restrict__ Wcb,
    const float* __restrict__ hbb, const float* __restrict__ cbb,
    float* __restrict__ dh, float* __restrict__ dc) {
  __shared__ _Float16 Ah[32 * 520];
  __shared__ _Float16 Ac[32 * 520];
  const int tid = threadIdx.x;
  const int lane = tid & 31, w = tid >> 5;
  const int lh = lane >> 4, jn = lane & 15;
  const int j = w * 16 + jn;
  for (int idx = tid; idx < 32 * 512; idx += 512) {
    const int m = idx >> 9, k = idx & 511;
    const int src = ((k >> 8) * 32 + m) * 256 + (k & 255);
    Ah[m * 520 + k] = (_Float16)hfin[src];
    Ac[m * 520 + k] = (_Float16)cfin[src];
  }
  __syncthreads();
  v8f ah[2] = {vzero8(), vzero8()};
  v8f ac[2] = {vzero8(), vzero8()};
  for (int kt = 0; kt < 16; ++kt) {
    v16h bh = bfrag(Whb, w * 16 + kt);
    v16h bc = bfrag(Wcb, w * 16 + kt);
    v16h a0 = afrag_lds(Ah, 520, 0, 0, kt);
    v16h a1 = afrag_lds(Ah, 520, 16, 0, kt);
    ah[0] = wmma_f16(a0, bh, ah[0]);
    ah[1] = wmma_f16(a1, bh, ah[1]);
    v16h e0 = afrag_lds(Ac, 520, 0, 0, kt);
    v16h e1 = afrag_lds(Ac, 520, 16, 0, kt);
    ac[0] = wmma_f16(e0, bc, ac[0]);
    ac[1] = wmma_f16(e1, bc, ac[1]);
  }
  const float bh_ = hbb[j], bc_ = cbb[j];
#pragma unroll
  for (int mt = 0; mt < 2; ++mt)
#pragma unroll
    for (int v = 0; v < 8; ++v) {
      const int m = mt * 16 + lh * 8 + v;
      dh[m * 256 + j] = ah[mt][v] + bh_;
      dc[m * 256 + j] = ac[mt][v] + bc_;
    }
}

// ---------------------------------------------------------------------------
// Autoregressive 2-layer decoder + fc, persistent single workgroup.
// A0 = [pred(24 pad 32) | h0] K=288 ; A1 = [h0' | h1] K=512.
// ---------------------------------------------------------------------------
__global__ void __launch_bounds__(512) dec_kernel(
    const _Float16* __restrict__ Wd0, const _Float16* __restrict__ Wd1,
    const _Float16* __restrict__ Wfc, const float* __restrict__ b0,
    const float* __restrict__ b1, const float* __restrict__ fcb,
    const float* __restrict__ dh, const float* __restrict__ dc,
    float* __restrict__ out) {
  __shared__ _Float16 A0[32 * 296];
  __shared__ _Float16 A1[32 * 520];
  const int tid = threadIdx.x;
  const int lane = tid & 31, w = tid >> 5;
  const int lh = lane >> 4, jn = lane & 15;
  const int j = w * 16 + jn;
  for (int idx = tid; idx < 32 * 256; idx += 512) {
    const int m = idx >> 8, k = idx & 255;
    const _Float16 v = (_Float16)dh[idx];
    A0[m * 296 + 32 + k] = v;    // h0 init = dh
    A1[m * 520 + 256 + k] = v;   // h1 init = dh
    A1[m * 520 + k] = (_Float16)0.f;
  }
  for (int idx = tid; idx < 32 * 32; idx += 512)
    A0[(idx >> 5) * 296 + (idx & 31)] = (_Float16)0.f;  // pred init = 0
  v8f c0[2], c1[2];
#pragma unroll
  for (int mt = 0; mt < 2; ++mt)
#pragma unroll
    for (int v = 0; v < 8; ++v) {
      const int m = mt * 16 + lh * 8 + v;
      c0[mt][v] = dc[m * 256 + j];
      c1[mt][v] = dc[m * 256 + j];
    }
  const float b0i = b0[j], b0f = b0[256 + j], b0g = b0[512 + j], b0o = b0[768 + j];
  const float b1i = b1[j], b1f = b1[256 + j], b1g = b1[512 + j], b1o = b1[768 + j];
  __syncthreads();
  for (int t = 0; t < 256; ++t) {
    float h0[2][8], h1[2][8];
    lstm_cell_step<9, 296>(A0, Wd0, b0i, b0f, b0g, b0o, c0, h0);
    __syncthreads();
#pragma unroll
    for (int mt = 0; mt < 2; ++mt)
#pragma unroll
      for (int v = 0; v < 8; ++v) {
        const int m = mt * 16 + lh * 8 + v;
        const _Float16 hh = (_Float16)h0[mt][v];
        A0[m * 296 + 32 + j] = hh;   // h0 recurrent input
        A1[m * 520 + j] = hh;        // layer-1 input
      }
    __syncthreads();
    lstm_cell_step<16, 520>(A1, Wd1, b1i, b1f, b1g, b1o, c1, h1);
    __syncthreads();
#pragma unroll
    for (int mt = 0; mt < 2; ++mt)
#pragma unroll
      for (int v = 0; v < 8; ++v) {
        const int m = mt * 16 + lh * 8 + v;
        A1[m * 520 + 256 + j] = (_Float16)h1[mt][v];
      }
    __syncthreads();
    // fc: pred = tanh(h1' @ fcW^T + fc_b), N padded 24->32, waves 0..3
    if (w < 4) {
      const int mt = w >> 1, nt = w & 1;
      v8f acc = vzero8();
      for (int kt = 0; kt < 8; ++kt) {
        v16h a = afrag_lds(A1, 520, mt * 16, 256, kt);
        v16h b = bfrag(Wfc, nt * 8 + kt);
        acc = wmma_f16(a, b, acc);
      }
      const int n = nt * 16 + jn;
      const float bb = (n < 24) ? fcb[n] : 0.f;
#pragma unroll
      for (int v = 0; v < 8; ++v) {
        const int m = mt * 16 + lh * 8 + v;
        const float pv = tanhf(acc[v] + bb);
        A0[m * 296 + n] = (n < 24) ? (_Float16)pv : (_Float16)0.f;
        if (n < 24) out[(m * 256 + t) * 24 + n] = pv;
      }
    }
    __syncthreads();
  }
}

// ---------------------------------------------------------------------------
// Prep kernels
// ---------------------------------------------------------------------------
// xs[bs][k] = f16(x[bs][k] * phylo[k]) for k<120, 0 for 120..127
__global__ void xs_kernel(const float* __restrict__ x,
                          const float* __restrict__ phylo,
                          _Float16* __restrict__ xs) {
  const int idx = blockIdx.x * 256 + threadIdx.x;
  if (idx >= 32 * 256 * 128) return;
  const int k = idx & 127, bs = idx >> 7;
  const float v = (k < 120) ? x[bs * 120 + k] * phylo[k] : 0.f;
  xs[idx] = (_Float16)v;
}

// Pack combined [A | pad | B] f32 weights into WMMA B-fragment tile layout:
// tile (nt,kt) -> 512 halves; half index = lane*16+e; lane holds
// W[n = nt*16 + (lane&15)][koff = kt*32 + (lane>>4)*8 + (e&7) + (e>>3)*16].
__global__ void pack_kernel(_Float16* __restrict__ dst,
                            const float* __restrict__ A,
                            const float* __restrict__ Bm, int NT, int KT,
                            int Nsrc, int Ka, int Kapad) {
  const int idx = blockIdx.x * 256 + threadIdx.x;
  const int total = NT * KT * 512;
  if (idx >= total) return;
  const int e = idx & 15;
  const int lane = (idx >> 4) & 31;
  const int tile = idx >> 9;
  const int kt = tile % KT, nt = tile / KT;
  const int n = nt * 16 + (lane & 15);
  const int koff = kt * 32 + ((lane >> 4) * 8) + (e & 7) + ((e >> 3) * 16);
  const int Ktot = KT * 32;
  float v = 0.f;
  if (n < Nsrc) {
    if (koff < Ka)
      v = A[n * Ka + koff];
    else if (koff >= Kapad && koff < Ktot)
      v = Bm[n * (Ktot - Kapad) + (koff - Kapad)];
  }
  dst[idx] = (_Float16)v;
}

__global__ void badd_kernel(float* __restrict__ dst, const float* __restrict__ a,
                            const float* __restrict__ b, int n) {
  const int i = blockIdx.x * 256 + threadIdx.x;
  if (i < n) dst[i] = a[i] + b[i];
}

// ---------------------------------------------------------------------------
// Host launch
// ---------------------------------------------------------------------------
extern "C" void kernel_launch(void* const* d_in, const int* in_sizes, int n_in,
                              void* d_out, int out_size, void* d_ws,
                              size_t ws_size, hipStream_t stream) {
  (void)in_sizes; (void)n_in; (void)out_size; (void)ws_size;
  const float* x = (const float*)d_in[0];
  const float* phylo = (const float*)d_in[2];
  // enc weights: base 3, order [l][dir]{Wih,Whh,bih,bhh}
  const float* eW[2][2][4];
  for (int l = 0; l < 2; ++l)
    for (int d = 0; d < 2; ++d)
      for (int k = 0; k < 4; ++k) eW[l][d][k] = (const float*)d_in[3 + l * 8 + d * 4 + k];
  const float* hbW = (const float*)d_in[19];
  const float* hbB = (const float*)d_in[20];
  const float* cbW = (const float*)d_in[21];
  const float* cbB = (const float*)d_in[22];
  const float* dW[2][4];
  for (int l = 0; l < 2; ++l)
    for (int k = 0; k < 4; ++k) dW[l][k] = (const float*)d_in[23 + l * 4 + k];
  const float* fcW = (const float*)d_in[31];
  const float* fcB = (const float*)d_in[32];
  float* out = (float*)d_out;

  char* p = (char*)d_ws;
  auto carve = [&](size_t bytes) {
    char* r = p;
    p += (bytes + 255) & ~(size_t)255;
    return r;
  };
  _Float16* xs = (_Float16*)carve((size_t)32 * 256 * 128 * 2);
  _Float16* Wl0[2] = {(_Float16*)carve((size_t)64 * 12 * 512 * 2),
                      (_Float16*)carve((size_t)64 * 12 * 512 * 2)};
  _Float16* Wl1[2] = {(_Float16*)carve((size_t)64 * 24 * 512 * 2),
                      (_Float16*)carve((size_t)64 * 24 * 512 * 2)};
  _Float16* Wd0t = (_Float16*)carve((size_t)64 * 9 * 512 * 2);
  _Float16* Wd1t = (_Float16*)carve((size_t)64 * 16 * 512 * 2);
  _Float16* Whbt = (_Float16*)carve((size_t)16 * 16 * 512 * 2);
  _Float16* Wcbt = (_Float16*)carve((size_t)16 * 16 * 512 * 2);
  _Float16* Wfct = (_Float16*)carve((size_t)2 * 8 * 512 * 2);
  float* bl0[2] = {(float*)carve(1024 * 4), (float*)carve(1024 * 4)};
  float* bl1[2] = {(float*)carve(1024 * 4), (float*)carve(1024 * 4)};
  float* bd0 = (float*)carve(1024 * 4);
  float* bd1 = (float*)carve(1024 * 4);
  _Float16* o0 = (_Float16*)carve((size_t)32 * 256 * 512 * 2);
  float* h1fin = (float*)carve((size_t)2 * 32 * 256 * 4);
  float* c1fin = (float*)carve((size_t)2 * 32 * 256 * 4);
  float* dhB = (float*)carve((size_t)32 * 256 * 4);
  float* dcB = (float*)carve((size_t)32 * 256 * 4);

  auto gb = [](int n) { return (n + 255) / 256; };

  xs_kernel<<<gb(32 * 256 * 128), 256, 0, stream>>>(x, phylo, xs);
  // enc l0 f/b: [Wih(120 pad 128) | Whh(256)], K=384
  for (int d = 0; d < 2; ++d) {
    pack_kernel<<<gb(64 * 12 * 512), 256, 0, stream>>>(
        Wl0[d], eW[0][d][0], eW[0][d][1], 64, 12, 1024, 120, 128);
    pack_kernel<<<gb(64 * 24 * 512), 256, 0, stream>>>(
        Wl1[d], eW[1][d][0], eW[1][d][1], 64, 24, 1024, 512, 512);
    badd_kernel<<<gb(1024), 256, 0, stream>>>(bl0[d], eW[0][d][2], eW[0][d][3], 1024);
    badd_kernel<<<gb(1024), 256, 0, stream>>>(bl1[d], eW[1][d][2], eW[1][d][3], 1024);
  }
  pack_kernel<<<gb(64 * 9 * 512), 256, 0, stream>>>(Wd0t, dW[0][0], dW[0][1], 64, 9, 1024, 24, 32);
  pack_kernel<<<gb(64 * 16 * 512), 256, 0, stream>>>(Wd1t, dW[1][0], dW[1][1], 64, 16, 1024, 256, 256);
  pack_kernel<<<gb(16 * 16 * 512), 256, 0, stream>>>(Whbt, hbW, nullptr, 16, 16, 256, 512, 512);
  pack_kernel<<<gb(16 * 16 * 512), 256, 0, stream>>>(Wcbt, cbW, nullptr, 16, 16, 256, 512, 512);
  pack_kernel<<<gb(2 * 8 * 512), 256, 0, stream>>>(Wfct, fcW, nullptr, 2, 8, 24, 256, 256);
  badd_kernel<<<gb(1024), 256, 0, stream>>>(bd0, dW[0][2], dW[0][3], 1024);
  badd_kernel<<<gb(1024), 256, 0, stream>>>(bd1, dW[1][2], dW[1][3], 1024);

  // encoder layer 0 (both directions concurrently), writes o0 = [f|b] f16
  lstm_enc_kernel<12, 392, 128, true, false><<<2, 512, 0, stream>>>(
      xs, Wl0[0], Wl0[1], bl0[0], bl0[1], o0, nullptr, nullptr);
  // encoder layer 1: only final h,c
  lstm_enc_kernel<24, 776, 512, false, true><<<2, 512, 0, stream>>>(
      o0, Wl1[0], Wl1[1], bl1[0], bl1[1], nullptr, h1fin, c1fin);
  // bridge
  bridge_kernel<<<1, 512, 0, stream>>>(h1fin, c1fin, Whbt, Wcbt, hbB, cbB, dhB, dcB);
  // decoder
  dec_kernel<<<1, 512, 0, stream>>>(Wd0t, Wd1t, Wfct, bd0, bd1, fcB, dhB, dcB, out);
}
